// ReverseLSTMLayer_88115549045325
// MI455X (gfx1250) — matrix operand order repaired and checked by reading, not compiled, not run on hardware
//
#include <hip/hip_runtime.h>

#define SEQL 256
#define BATCH 64
#define IND 2048
#define HD 2048
#define G4 (4 * HD) // 8192

typedef __attribute__((ext_vector_type(16))) __bf16 v16bf;
typedef __attribute__((ext_vector_type(8)))  float  v8f;

union FragU { uint4 q[2]; v16bf v; };

__device__ __forceinline__ unsigned short f32_to_bf16_rne(float f) {
  unsigned x = __float_as_uint(f);
  return (unsigned short)((x + 0x7FFFu + ((x >> 16) & 1u)) >> 16);
}

// A-matrix 16x32 bf16 fragment (ISA 7.12.2 layout):
// lanes 0-15 row M=lane, elems = K[k0..k0+7] ++ K[k0+16..k0+23]
// lanes 16-31 row M=lane-16, elems = K[k0+8..k0+15] ++ K[k0+24..k0+31]
__device__ __forceinline__ v16bf load_a_frag(const unsigned short* __restrict__ base,
                                             int ld, int row0, int k0, int lane) {
  const int r  = row0 + (lane & 15);
  const int kb = k0 + ((lane >> 4) << 3);
  const unsigned short* p = base + (size_t)r * ld + kb;
  FragU u;
  u.q[0] = *(const uint4*)(p);       // 8 bf16: K = kb .. kb+7
  u.q[1] = *(const uint4*)(p + 16);  // 8 bf16: K = kb+16 .. kb+23
  return u.v;
}

// B-matrix 32x16 bf16 fragment: lane holds column N = lane&15;
// lanes 0-15 hold K[k0..k0+7] ++ K[k0+16..k0+23], lanes 16-31 hold K[k0+8..k0+15] ++ K[k0+24..k0+31].
// B(k,n) = W[n_global, k] with W row-major [G4, K].
__device__ __forceinline__ v16bf load_b_frag(const unsigned short* __restrict__ W,
                                             int ld, int n0, int k0, int lane) {
  const int n  = n0 + (lane & 15);
  const int kb = k0 + ((lane >> 4) << 3);
  const unsigned short* p = W + (size_t)n * ld + kb;
  FragU u;
  u.q[0] = *(const uint4*)(p);
  u.q[1] = *(const uint4*)(p + 16);
  return u.v;
}

__device__ __forceinline__ float fast_sigmoid(float x) {
  return 1.0f / (1.0f + __expf(-x));
}
__device__ __forceinline__ float fast_tanh(float x) {
  float e = __expf(2.0f * x);
  return (e - 1.0f) / (e + 1.0f);
}

// One reverse-time LSTM step. grid = 128 blocks (16-col group per block,
// same columns across all 4 gates), block = 256 threads = 8 waves (split-K=8).
__global__ void __launch_bounds__(256) lstm_step_kernel(
    const unsigned short* __restrict__ Xbf,   // [SEQL*BATCH, IND] bf16
    const unsigned short* __restrict__ Wihbf, // [G4, IND] bf16 row-major
    const unsigned short* __restrict__ Whhbf, // [G4, HD]  bf16 row-major
    const float* __restrict__ bias,           // [G4] = b_ih + b_hh
    const unsigned short* __restrict__ h_in,  // [BATCH, HD] bf16
    unsigned short* __restrict__ h_out,       // [BATCH, HD] bf16
    float* __restrict__ cbuf,                 // [BATCH, HD] f32 (in/out)
    float* __restrict__ outs,                 // [SEQL, BATCH, HD] f32
    int t) {
  __shared__ float gl[4 * BATCH * 16]; // [gate][row b][col] = 16 KB

  const int lane  = threadIdx.x & 31;
  const int wave  = threadIdx.x >> 5;   // 0..7 : K split
  const int ntile = blockIdx.x;         // 0..127 : col block within each gate

  for (int i = threadIdx.x; i < 4 * BATCH * 16; i += 256) gl[i] = 0.0f;
  __syncthreads();

  v8f acc[4][4] = {}; // [gate][mtile], f32 16x16 accumulators

  const int kbeg = wave * (IND / 8); // 256-wide K chunk per wave

  // ---- x_t @ W_ih^T ----
  {
    const unsigned short* Xs = Xbf + (size_t)t * BATCH * IND;
    for (int k0 = kbeg; k0 < kbeg + IND / 8; k0 += 32) {
      v16bf a[4];
#pragma unroll
      for (int mt = 0; mt < 4; ++mt) a[mt] = load_a_frag(Xs, IND, mt * 16, k0, lane);
#pragma unroll
      for (int g = 0; g < 4; ++g) {
        v16bf b = load_b_frag(Wihbf, IND, g * HD + ntile * 16, k0, lane);
#pragma unroll
        for (int mt = 0; mt < 4; ++mt)
          acc[g][mt] = __builtin_amdgcn_wmma_f32_16x16x32_bf16(
              false, a[mt], false, b, (short)0, acc[g][mt], false, false);
      }
    }
  }

  // ---- h @ W_hh^T ----
  {
    for (int k0 = kbeg; k0 < kbeg + HD / 8; k0 += 32) {
      v16bf a[4];
#pragma unroll
      for (int mt = 0; mt < 4; ++mt) a[mt] = load_a_frag(h_in, HD, mt * 16, k0, lane);
#pragma unroll
      for (int g = 0; g < 4; ++g) {
        v16bf b = load_b_frag(Whhbf, HD, g * HD + ntile * 16, k0, lane);
#pragma unroll
        for (int mt = 0; mt < 4; ++mt)
          acc[g][mt] = __builtin_amdgcn_wmma_f32_16x16x32_bf16(
              false, a[mt], false, b, (short)0, acc[g][mt], false, false);
      }
    }
  }

  // ---- split-K reduction into LDS (ds_add_f32) ----
  // C/D layout: lanes 0-15 -> M=v, N=lane; lanes 16-31 -> M=v+8, N=lane-16
#pragma unroll
  for (int g = 0; g < 4; ++g)
#pragma unroll
    for (int mt = 0; mt < 4; ++mt)
#pragma unroll
      for (int v = 0; v < 8; ++v) {
        int row = mt * 16 + ((lane >> 4) << 3) + v;
        int col = lane & 15;
        atomicAdd(&gl[(g * BATCH + row) * 16 + col], acc[g][mt][v]);
      }
  __syncthreads();

  // ---- fused LSTM cell epilogue ----
  for (int idx = threadIdx.x; idx < BATCH * 16; idx += 256) {
    int b = idx >> 4;
    int n = idx & 15;
    int col = ntile * 16 + n;
    float xi = gl[(0 * BATCH + b) * 16 + n] + bias[0 * HD + col];
    float xf = gl[(1 * BATCH + b) * 16 + n] + bias[1 * HD + col];
    float xg = gl[(2 * BATCH + b) * 16 + n] + bias[2 * HD + col];
    float xo = gl[(3 * BATCH + b) * 16 + n] + bias[3 * HD + col];
    float i_ = fast_sigmoid(xi);
    float f_ = fast_sigmoid(xf);
    float g_ = fast_tanh(xg);
    float o_ = fast_sigmoid(xo);
    size_t p = (size_t)b * HD + col;
    float cn = f_ * cbuf[p] + i_ * g_;
    float hn = o_ * fast_tanh(cn);
    cbuf[p] = cn;
    outs[((size_t)t * BATCH + b) * HD + col] = hn;
    h_out[p] = f32_to_bf16_rne(hn);
  }
}

__global__ void cvt_f32_bf16_kernel(const float* __restrict__ src,
                                    unsigned short* __restrict__ dst, int n) {
  int i = blockIdx.x * blockDim.x + threadIdx.x;
  if (i < n) dst[i] = f32_to_bf16_rne(src[i]);
}

__global__ void bias_sum_kernel(const float* __restrict__ a, const float* __restrict__ b,
                                float* __restrict__ o, int n) {
  int i = blockIdx.x * blockDim.x + threadIdx.x;
  if (i < n) o[i] = a[i] + b[i];
}

// h_T = outs[0] (reverse scan's last-produced h), c_T from state buffer.
__global__ void tail_kernel(const float* __restrict__ outs, const float* __restrict__ cbuf,
                            float* __restrict__ dst_h, float* __restrict__ dst_c, int n) {
  int i = blockIdx.x * blockDim.x + threadIdx.x;
  if (i < n) {
    dst_h[i] = outs[i];
    dst_c[i] = cbuf[i];
  }
}

extern "C" void kernel_launch(void* const* d_in, const int* in_sizes, int n_in,
                              void* d_out, int out_size, void* d_ws, size_t ws_size,
                              hipStream_t stream) {
  const float* X   = (const float*)d_in[0]; // [SEQL, BATCH, IND]
  const float* h0  = (const float*)d_in[1]; // [BATCH, HD]
  const float* c0  = (const float*)d_in[2]; // [BATCH, HD]
  const float* Wih = (const float*)d_in[3]; // [G4, IND]
  const float* Whh = (const float*)d_in[4]; // [G4, HD]
  const float* bih = (const float*)d_in[5]; // [G4]
  const float* bhh = (const float*)d_in[6]; // [G4]
  float* out = (float*)d_out;

  char* ws = (char*)d_ws;
  const size_t XBF_OFF  = 0;                                   // 67,108,864 B
  const size_t WIH_OFF  = XBF_OFF + (size_t)SEQL * BATCH * IND * 2;
  const size_t WHH_OFF  = WIH_OFF + (size_t)G4 * IND * 2;      // +33,554,432
  const size_t HBF0_OFF = WHH_OFF + (size_t)G4 * HD * 2;       // +33,554,432
  const size_t HBF1_OFF = HBF0_OFF + (size_t)BATCH * HD * 2;
  const size_t C_OFF    = HBF1_OFF + (size_t)BATCH * HD * 2;
  const size_t BIAS_OFF = C_OFF + (size_t)BATCH * HD * 4;

  unsigned short* Xbf    = (unsigned short*)(ws + XBF_OFF);
  unsigned short* Wihbf  = (unsigned short*)(ws + WIH_OFF);
  unsigned short* Whhbf  = (unsigned short*)(ws + WHH_OFF);
  unsigned short* hbf[2] = {(unsigned short*)(ws + HBF0_OFF),
                            (unsigned short*)(ws + HBF1_OFF)};
  float* cbuf = (float*)(ws + C_OFF);
  float* bias = (float*)(ws + BIAS_OFF);

  // ---- per-call setup: bf16 conversions, bias fold, state init ----
  {
    int n;
    n = SEQL * BATCH * IND;
    cvt_f32_bf16_kernel<<<(n + 255) / 256, 256, 0, stream>>>(X, Xbf, n);
    n = G4 * IND;
    cvt_f32_bf16_kernel<<<(n + 255) / 256, 256, 0, stream>>>(Wih, Wihbf, n);
    n = G4 * HD;
    cvt_f32_bf16_kernel<<<(n + 255) / 256, 256, 0, stream>>>(Whh, Whhbf, n);
    n = BATCH * HD;
    cvt_f32_bf16_kernel<<<(n + 255) / 256, 256, 0, stream>>>(h0, hbf[0], n);
    n = G4;
    bias_sum_kernel<<<(n + 255) / 256, 256, 0, stream>>>(bih, bhh, bias, n);
    hipMemcpyAsync(cbuf, c0, (size_t)BATCH * HD * sizeof(float),
                   hipMemcpyDeviceToDevice, stream);
  }

  // ---- reverse-time recurrence: 256 dependent step kernels ----
  int ping = 0;
  for (int t = SEQL - 1; t >= 0; --t) {
    lstm_step_kernel<<<G4 / 16 / 4, 256, 0, stream>>>(
        Xbf, Wihbf, Whhbf, bias, hbf[ping], hbf[1 - ping], cbuf, out, t);
    ping ^= 1;
  }

  // ---- append (h_T, c_T) after outs ----
  {
    int n = BATCH * HD;
    float* dst_h = out + (size_t)SEQL * BATCH * HD;
    float* dst_c = dst_h + (size_t)BATCH * HD;
    tail_kernel<<<(n + 255) / 256, 256, 0, stream>>>(out, cbuf, dst_h, dst_c, n);
  }
}
